// VectorQuantizer_31894427140465
// MI455X (gfx1250) — matrix operand order repaired and checked by reading, not compiled, not run on hardware
//
#include <hip/hip_runtime.h>
#include <hip/hip_bf16.h>

// Problem constants (match reference).
#define BDIM 16
#define DDIM 256
#define TDIM 2048
#define KEMB 1024
#define NROW (BDIM * TDIM)          // 32768 flattened vectors
#define QN   (BDIM * DDIM * TDIM)   // 8388608 quantized elements

#define EPITCH 264                  // LDS row pitch (elements): 528B -> conflict-free

typedef __bf16 v16bf __attribute__((ext_vector_type(16)));
typedef __bf16 v8bf  __attribute__((ext_vector_type(8)));
typedef float  v8f   __attribute__((ext_vector_type(8)));
typedef int    v4i   __attribute__((ext_vector_type(4)));

// ---- CDNA5 async global->LDS copy (ASYNCcnt path), with sync fallback ------
// Probe-confirmed signature: (v4i addrspace(1)*, v4i addrspace(3)*, imm, imm)
#if __has_builtin(__builtin_amdgcn_global_load_async_to_lds_b128)
#define VQ_ASYNC_LDS 1
__device__ __forceinline__ void async_copy16(const unsigned short* g, unsigned short* l) {
    __builtin_amdgcn_global_load_async_to_lds_b128(
        (__attribute__((address_space(1))) v4i*)g,
        (__attribute__((address_space(3))) v4i*)l, 0, 0);
}
__device__ __forceinline__ void wait_async() {
#if __has_builtin(__builtin_amdgcn_s_wait_asynccnt)
    __builtin_amdgcn_s_wait_asynccnt(0);
#else
    asm volatile("s_wait_asynccnt 0" ::: "memory");
#endif
}
#else
__device__ __forceinline__ void async_copy16(const unsigned short* g, unsigned short* l) {
    *reinterpret_cast<v8bf*>(l) = *reinterpret_cast<const v8bf*>(g);
}
__device__ __forceinline__ void wait_async() {}
#endif

// ---- bf16 helpers (round-to-nearest-even via bit ops; storage = ushort) ----
__device__ __forceinline__ unsigned short f2bf(float f) {
    unsigned int u = __float_as_uint(f);
    u += 0x7FFFu + ((u >> 16) & 1u);
    return (unsigned short)(u >> 16);
}
__device__ __forceinline__ float bf2f(unsigned short h) {
    return __uint_as_float(((unsigned int)h) << 16);
}

// ---- WMMA fragment loads (16-bit A/B VGPR striping per CDNA5 ISA 7.12.2) ---
// A (16x32, M x K): lane holds row M = lane&15; K chunks {k0..k0+7},{k0+16..k0+23},
// k0 = (lane>>4)*8.
__device__ __forceinline__ v16bf load_a_frag(const unsigned short* rowptr, int lane) {
    int k0 = (lane >> 4) << 3;
    v8bf x0 = *reinterpret_cast<const v8bf*>(rowptr + k0);
    v8bf x1 = *reinterpret_cast<const v8bf*>(rowptr + k0 + 16);
    return __builtin_shufflevector(x0, x1, 0, 1, 2, 3, 4, 5, 6, 7,
                                   8, 9, 10, 11, 12, 13, 14, 15);
}
// B (32x16, K x N): lane holds column N = lane&15; K chunks {k0..k0+7},{k0+16..k0+23}, k0 = (lane>>4)*8.
__device__ __forceinline__ v16bf load_b_frag(const unsigned short* colptr, int lane) {
    int k0 = (lane >> 4) << 3;
    v8bf x0 = *reinterpret_cast<const v8bf*>(colptr + k0);
    v8bf x1 = *reinterpret_cast<const v8bf*>(colptr + k0 + 16);
    return __builtin_shufflevector(x0, x1, 0, 1, 2, 3, 4, 5, 6, 7,
                                   8, 9, 10, 11, 12, 13, 14, 15);
}

__device__ __forceinline__ v8f wmma_bf16(v16bf a, v16bf b, v8f c) {
    // (neg_a, A, neg_b, B, c_mod, C, reuse_a, reuse_b)
    return __builtin_amdgcn_wmma_f32_16x16x32_bf16(false, a, false, b,
                                                   (short)0, c, false, false);
}

// ============================================================================
// Kernel 1: transpose [B, D, T] -> row-major [N=B*T, D] with bf16 hi/lo split.
// Grid: B * (T/32) = 1024 blocks, 256 threads. LDS 32x32 tile, coalesced both ways.
// ============================================================================
__global__ void vq_prep_x(const float* __restrict__ x,
                          unsigned short* __restrict__ x_hi,
                          unsigned short* __restrict__ x_lo) {
    __shared__ float tile[32][33];
    const int b  = blockIdx.x >> 6;          // T/32 = 64 tiles per batch
    const int t0 = (blockIdx.x & 63) * 32;
    const int tid = threadIdx.x;

    for (int d0 = 0; d0 < DDIM; d0 += 32) {
        {   // Load phase: t fastest across lanes (coalesced reads).
            const int tl = tid & 31;
            const int dl = tid >> 5;         // 0..7
            #pragma unroll
            for (int i = 0; i < 4; ++i) {
                int d = dl + 8 * i;
                tile[d][tl] = x[((size_t)b * DDIM + d0 + d) * TDIM + t0 + tl];
            }
        }
        __syncthreads();
        {   // Store phase: d fastest across lanes (coalesced bf16 writes).
            const int dw = tid & 31;
            const int tw = tid >> 5;         // 0..7
            #pragma unroll
            for (int i = 0; i < 4; ++i) {
                int t = tw + 8 * i;
                float v = tile[dw][t];
                unsigned short h = f2bf(v);
                unsigned short l = f2bf(v - bf2f(h));
                size_t n = (size_t)b * TDIM + t0 + t;
                x_hi[n * DDIM + d0 + dw] = h;
                x_lo[n * DDIM + d0 + dw] = l;
            }
        }
        __syncthreads();
    }
}

// ============================================================================
// Kernel 2: codebook hi/lo split (already [K, D] row-major). 1024x256 threads.
// ============================================================================
__global__ void vq_prep_e(const float* __restrict__ e,
                          unsigned short* __restrict__ e_hi,
                          unsigned short* __restrict__ e_lo) {
    int i = blockIdx.x * blockDim.x + threadIdx.x;   // covers K*D = 262144
    float v = e[i];
    unsigned short h = f2bf(v);
    e_hi[i] = h;
    e_lo[i] = f2bf(v - bf2f(h));
}

// ============================================================================
// Kernel 3: per-embedding squared norms. 128 blocks x 8 waves, 1 wave per row.
// ============================================================================
__global__ void vq_enorm(const float* __restrict__ e, float* __restrict__ en) {
    const int lane = threadIdx.x & 31;
    const int wave = threadIdx.x >> 5;
    const int k = blockIdx.x * 8 + wave;
    float s = 0.0f;
    #pragma unroll
    for (int i = 0; i < 8; ++i) {
        float v = e[k * DDIM + lane + 32 * i];
        s += v * v;
    }
    #pragma unroll
    for (int off = 16; off >= 1; off >>= 1)
        s += __shfl_xor(s, off, 32);
    if (lane == 0) en[k] = s;
}

// ============================================================================
// Kernel 4: distance GEMM + running argmin, bf16x3 WMMA (fp32-accurate dots).
// 256 blocks x 256 threads (8 waves); each wave owns a 16-row M-tile.
// Codebook tiles (16 emb x 256 k, hi+lo) are double-buffered in LDS via
// async global->LDS copies; A-hi lives in registers, A-lo streams from cache.
// d2 = |e|^2 - 2 * (x . e)   (|x|^2 dropped: argmin-invariant)
// ============================================================================
__device__ __forceinline__ void stage_tile(const unsigned short* __restrict__ e_hi,
                                           const unsigned short* __restrict__ e_lo,
                                           unsigned short (*dh)[EPITCH],
                                           unsigned short (*dl)[EPITCH],
                                           int et, int tid) {
    // 16 rows x 32 chunks (16B) for hi and for lo = 1024 chunks, 4 per thread.
    #pragma unroll
    for (int i = 0; i < 4; ++i) {
        int c   = tid + 256 * i;             // 0..1023
        int cc  = c & 511;
        int row = cc >> 5;                   // 0..15
        int col = (cc & 31) * 8;             // element offset (16B chunks)
        const unsigned short* gsrc =
            (i < 2 ? e_hi : e_lo) + ((size_t)(et * 16 + row) * DDIM + col);
        unsigned short* ldst = (i < 2 ? dh : dl)[row] + col;
        async_copy16(gsrc, ldst);
    }
}

__global__ __launch_bounds__(256, 1) void vq_dist(
        const unsigned short* __restrict__ x_hi,
        const unsigned short* __restrict__ x_lo,
        const unsigned short* __restrict__ e_hi,
        const unsigned short* __restrict__ e_lo,
        const float* __restrict__ e_norm,
        int* __restrict__ indices) {
    __shared__ __align__(16) unsigned short ls_eh[2][16][EPITCH];
    __shared__ __align__(16) unsigned short ls_el[2][16][EPITCH];

    const int tid  = threadIdx.x;
    const int lane = tid & 31;
    const int wave = tid >> 5;
    const int rowbase = blockIdx.x * 128 + wave * 16;
    const int myrow   = rowbase + (lane & 15);

    const unsigned short* xh = x_hi + (size_t)myrow * DDIM;
    const unsigned short* xl = x_lo + (size_t)myrow * DDIM;

    // Persistent A-hi fragments: 8 x v16bf = 64 VGPRs.
    v16bf ahi[8];
    #pragma unroll
    for (int ks = 0; ks < DDIM / 32; ++ks)
        ahi[ks] = load_a_frag(xh + ks * 32, lane);

    float minv[8];
    int   mini[8];
    #pragma unroll
    for (int j = 0; j < 8; ++j) { minv[j] = 3.4e38f; mini[j] = 0; }

    stage_tile(e_hi, e_lo, ls_eh[0], ls_el[0], 0, tid);

    for (int et = 0; et < KEMB / 16; ++et) {
        const int p = et & 1;
        wait_async();
        __syncthreads();
        if (et + 1 < KEMB / 16)
            stage_tile(e_hi, e_lo, ls_eh[p ^ 1], ls_el[p ^ 1], et + 1, tid);

        const int ncol = et * 16 + (lane & 15);
        const unsigned short* brow_h = &ls_eh[p][lane & 15][0];
        const unsigned short* brow_l = &ls_el[p][lane & 15][0];

        v8f acc = {0.f, 0.f, 0.f, 0.f, 0.f, 0.f, 0.f, 0.f};
        #pragma unroll
        for (int ks = 0; ks < DDIM / 32; ++ks) {
            v16bf alo = load_a_frag(xl + ks * 32, lane);       // streams from L1
            v16bf bh  = load_b_frag(brow_h + ks * 32, lane);   // ds_load_b128
            v16bf bl  = load_b_frag(brow_l + ks * 32, lane);
            acc = wmma_bf16(ahi[ks], bh, acc);  // hi*hi
            acc = wmma_bf16(ahi[ks], bl, acc);  // hi*lo
            acc = wmma_bf16(alo,     bh, acc);  // lo*hi
        }

        const float en = e_norm[ncol];
        #pragma unroll
        for (int j = 0; j < 8; ++j) {
            float d2 = en - 2.0f * acc[j];
            bool take = (d2 < minv[j]) || (d2 == minv[j] && ncol < mini[j]);
            if (take) { minv[j] = d2; mini[j] = ncol; }
        }
        __syncthreads();   // tile p fully consumed before it is restaged
    }

    // Min-reduce over the 16 lanes holding the same 8 rows (xor stays in half-wave).
    #pragma unroll
    for (int off = 1; off < 16; off <<= 1) {
        #pragma unroll
        for (int j = 0; j < 8; ++j) {
            float ov = __shfl_xor(minv[j], off, 32);
            int   oi = __shfl_xor(mini[j], off, 32);
            bool take = (ov < minv[j]) || (ov == minv[j] && oi < mini[j]);
            if (take) { minv[j] = ov; mini[j] = oi; }
        }
    }
    if ((lane & 15) == 0) {
        int mrow = (lane >> 4) * 8;   // lane 0 -> rows 0..7, lane 16 -> rows 8..15
        #pragma unroll
        for (int j = 0; j < 8; ++j)
            indices[rowbase + mrow + j] = mini[j];
    }
}

// ============================================================================
// Kernel 5: gather quantized output [B, D, T], per-block SSE partials,
// indices-as-float into output tail. Grid 512 blocks (b x 64-t tile), 256 thr.
// ============================================================================
__global__ void vq_gather(const float* __restrict__ x,
                          const float* __restrict__ emb,
                          const int* __restrict__ idx,
                          float* __restrict__ out,
                          float* __restrict__ partials) {
    __shared__ float red[256];
    const int tid  = threadIdx.x;
    const int b    = blockIdx.x >> 5;          // T/64 = 32 tiles per batch
    const int t0   = (blockIdx.x & 31) * 64;
    const int toff = tid & 63;
    const int dgrp = tid >> 6;                 // 0..3
    const int t    = t0 + toff;
    const int n    = b * TDIM + t;
    const int k    = idx[n];

    float s = 0.0f;
    for (int d = dgrp; d < DDIM; d += 4) {
        float q  = emb[k * DDIM + d];
        size_t o = ((size_t)b * DDIM + d) * TDIM + t;
        float xv = x[o];
        out[o] = q;                            // straight-through == quantized
        float df = q - xv;
        s += df * df;
    }
    red[tid] = s;
    __syncthreads();
    for (int stride = 128; stride >= 1; stride >>= 1) {
        if (tid < stride) red[tid] += red[tid + stride];
        __syncthreads();
    }
    if (tid == 0) partials[blockIdx.x] = red[0];

    if (tid < 64) {
        int tt = t0 + tid;
        int nn = b * TDIM + tt;
        out[QN + 3 + nn] = (float)idx[nn];
    }
}

// ============================================================================
// Kernel 6: deterministic fixed-tree reduction of 512 partials -> loss scalars.
// ============================================================================
__global__ void vq_final(const float* __restrict__ partials, float* __restrict__ out) {
    __shared__ float red[256];
    const int tid = threadIdx.x;
    red[tid] = partials[tid] + partials[tid + 256];
    __syncthreads();
    for (int stride = 128; stride >= 1; stride >>= 1) {
        if (tid < stride) red[tid] += red[tid + stride];
        __syncthreads();
    }
    if (tid == 0) {
        float mse = red[0] / (float)QN;
        out[QN + 0] = 1.25f * mse;   // q_latent + 0.25 * e_latent
        out[QN + 1] = mse;           // e_latent_loss
        out[QN + 2] = mse;           // q_latent_loss
    }
}

// ============================================================================
extern "C" void kernel_launch(void* const* d_in, const int* in_sizes, int n_in,
                              void* d_out, int out_size, void* d_ws, size_t ws_size,
                              hipStream_t stream) {
    const float* x   = (const float*)d_in[0];   // [B, D, T] fp32
    const float* emb = (const float*)d_in[1];   // [K, D]    fp32
    float* out = (float*)d_out;

    // Workspace layout (byte offsets, all 16B-aligned):
    char* ws = (char*)d_ws;
    unsigned short* x_hi   = (unsigned short*)(ws);                                  // 16 MB
    unsigned short* x_lo   = (unsigned short*)(ws + 16777216);                       // 16 MB
    unsigned short* e_hi   = (unsigned short*)(ws + 2 * 16777216);                   // 512 KB
    unsigned short* e_lo   = (unsigned short*)(ws + 2 * 16777216 + 524288);          // 512 KB
    float*          e_nrm  = (float*)(ws + 2 * 16777216 + 2 * 524288);               // 4 KB
    int*            indic  = (int*)(ws + 2 * 16777216 + 2 * 524288 + 4096);          // 128 KB
    float*          parts  = (float*)(ws + 2 * 16777216 + 2 * 524288 + 4096 + 131072); // 2 KB

    vq_prep_x<<<BDIM * (TDIM / 32), 256, 0, stream>>>(x, x_hi, x_lo);
    vq_prep_e<<<(KEMB * DDIM) / 256, 256, 0, stream>>>(emb, e_hi, e_lo);
    vq_enorm<<<KEMB / 8, 256, 0, stream>>>(emb, e_nrm);
    vq_dist<<<NROW / 128, 256, 0, stream>>>(x_hi, x_lo, e_hi, e_lo, e_nrm, indic);
    vq_gather<<<BDIM * (TDIM / 64), 256, 0, stream>>>(x, emb, indic, out, parts);
    vq_final<<<1, 256, 0, stream>>>(parts, out);
}
